// AdaptiveTemporalVQ_56882546868551
// MI455X (gfx1250) — compile-verified, matrix operand order checked by reading, not run on hardware
//
#include <hip/hip_runtime.h>

#define NUM_EMB_K 8192
#define EMB_D     512
#define SPAN_SZ   8
#define BATCH     16
#define T_LEN     8192
#define S_LEN     (T_LEN / SPAN_SZ)     // 1024
#define N_SEG     (BATCH * S_LEN)       // 16384
#define BT        (BATCH * T_LEN)       // 131072

// d_out layout (floats), tuple order: quantized, loss, idx, boundaries
#define OUT_Q     ((size_t)0)
#define OUT_LOSS  ((size_t)BATCH * T_LEN * EMB_D)          // 67,108,864
#define OUT_IDX   (OUT_LOSS + 1)
#define OUT_BND   (OUT_IDX + (size_t)BT)

// ws layout (float units)
#define WS_EMBNORM ((size_t)0)                                    // 8192 f32
#define WS_IDX     ((size_t)8192)                                 // 16384 i32
#define WS_EMBBF   ((size_t)24576)                                // 8192*512 bf16
#define WS_POOLED  (WS_EMBBF + (size_t)NUM_EMB_K * EMB_D / 2)     // 16384*512 f32
#define WS_BPART   (WS_POOLED + (size_t)N_SEG * EMB_D)            // 16384 f32
#define WS_EPART   (WS_BPART + (size_t)N_SEG)                     // 16384 f32

typedef __attribute__((ext_vector_type(16))) __bf16 v16bf;
typedef __attribute__((ext_vector_type(8)))  __bf16 v8bf;
typedef __attribute__((ext_vector_type(8)))  float  v8f;
typedef __attribute__((ext_vector_type(4)))  int    v4i;

// ---- gfx1250 async global->LDS path (guarded; fallback = sync copy) -------
#if defined(__has_builtin)
#if __has_builtin(__builtin_amdgcn_global_load_async_to_lds_b128)
#define USE_ASYNC_LDS 1
#endif
#endif

#ifdef USE_ASYNC_LDS
typedef __attribute__((address_space(1))) v4i gas_v4i;   // global
typedef __attribute__((address_space(3))) v4i las_v4i;   // LDS

__device__ __forceinline__ void async_copy16(const __bf16* g, __bf16* l) {
  __builtin_amdgcn_global_load_async_to_lds_b128(
      (gas_v4i*)(unsigned long long)g,
      (las_v4i*)(unsigned)(unsigned long long)l,
      0, 0);
}
#endif

__device__ __forceinline__ void wait_async_lds() {
#ifdef USE_ASYNC_LDS
#if __has_builtin(__builtin_amdgcn_s_wait_asynccnt)
  __builtin_amdgcn_s_wait_asynccnt(0);
#else
  asm volatile("s_wait_asynccnt 0x0" ::: "memory");
#endif
#endif
}

// ---------------------------------------------------------------------------
// Kernel 1: fused boundary predictor + segment mean pooling (single x pass)
// grid = N_SEG blocks, 256 threads (2 dims/thread)
// ---------------------------------------------------------------------------
__global__ __launch_bounds__(256) void prep_kernel(
    const float* __restrict__ x, const float* __restrict__ Wb,
    const float* __restrict__ bb, float* __restrict__ out,
    float* __restrict__ ws) {
  __shared__ float red[SPAN_SZ * 256];
  const int tid = threadIdx.x;
  const int n = blockIdx.x;
  const int b = n / S_LEN, s = n % S_LEN;
  const int d0 = tid * 2;
  const size_t base = ((size_t)(b * T_LEN + s * SPAN_SZ)) * EMB_D + d0;
  const float2 wv = *(const float2*)(Wb + d0);
  float px = 0.f, py = 0.f;
#pragma unroll
  for (int j = 0; j < SPAN_SZ; ++j) {
    float2 xv = *(const float2*)(x + base + (size_t)j * EMB_D);
    px += xv.x; py += xv.y;
    red[j * 256 + tid] = xv.x * wv.x + xv.y * wv.y;
  }
  __syncthreads();
  for (int off = 128; off >= 1; off >>= 1) {
    if (tid < off) {
#pragma unroll
      for (int j = 0; j < SPAN_SZ; ++j)
        red[j * 256 + tid] += red[j * 256 + tid + off];
    }
    __syncthreads();
  }
  float* pooled = ws + WS_POOLED + (size_t)n * EMB_D;
  *(float2*)(pooled + d0) = make_float2(px * 0.125f, py * 0.125f);
  // boundaries: sigmoid(l) > 0.5  <=>  l > 0
  if (tid < SPAN_SZ) {
    float logit = red[tid * 256] + bb[0];
    float flag = (logit > 0.0f) ? 1.0f : 0.0f;
    out[OUT_BND + (size_t)b * T_LEN + s * SPAN_SZ + tid] = flag;
    red[tid] = flag;
  }
  __syncthreads();
  if (tid == 0) {
    float c = 0.f;
#pragma unroll
    for (int j = 0; j < SPAN_SZ; ++j) c += red[j];
    ws[WS_BPART + n] = c;
  }
}

// ---------------------------------------------------------------------------
// Kernel 2: per-code squared norms + one-time f32->bf16 codebook copy.
// grid = NUM_EMB/8 blocks, 256 threads (8 waves, wave per code row)
// ---------------------------------------------------------------------------
__global__ __launch_bounds__(256) void embnorm_kernel(
    const float* __restrict__ emb, float* __restrict__ ws) {
  const int w = threadIdx.x >> 5, lane = threadIdx.x & 31;
  const int row = blockIdx.x * 8 + w;
  const float* er = emb + (size_t)row * EMB_D + lane * 16;
  float4 a = *(const float4*)(er);
  float4 b = *(const float4*)(er + 4);
  float4 c = *(const float4*)(er + 8);
  float4 d = *(const float4*)(er + 12);
  float s = a.x*a.x + a.y*a.y + a.z*a.z + a.w*a.w
          + b.x*b.x + b.y*b.y + b.z*b.z + b.w*b.w
          + c.x*c.x + c.y*c.y + c.z*c.z + c.w*c.w
          + d.x*d.x + d.y*d.y + d.z*d.z + d.w*d.w;
  __bf16* eb = ((__bf16*)(ws + WS_EMBBF)) + (size_t)row * EMB_D + lane * 16;
  v8bf lo = { (__bf16)a.x, (__bf16)a.y, (__bf16)a.z, (__bf16)a.w,
              (__bf16)b.x, (__bf16)b.y, (__bf16)b.z, (__bf16)b.w };
  v8bf hi = { (__bf16)c.x, (__bf16)c.y, (__bf16)c.z, (__bf16)c.w,
              (__bf16)d.x, (__bf16)d.y, (__bf16)d.z, (__bf16)d.w };
  *(v8bf*)(eb) = lo;
  *(v8bf*)(eb + 8) = hi;
#pragma unroll
  for (int m = 16; m >= 1; m >>= 1) s += __shfl_xor(s, m, 32);
  if (lane == 0) ws[WS_EMBNORM + row] = s;
}

// ---------------------------------------------------------------------------
// Kernel 3: bf16 WMMA distance GEMM + running argmin.
// 4 waves/block (128 thr); each wave owns 16 pooled rows (A in registers).
// Per iteration: 32 codes (two N-tiles, two independent WMMA accumulator
// chains), double-buffered LDS filled via async global->LDS loads overlapped
// with compute. grid = N_SEG/64 = 256 blocks.
// ---------------------------------------------------------------------------
#define SB_STRIDE 520   // 16B-aligned, bank-conflict-free row stride (bf16)
#define NT_ITERS  (NUM_EMB_K / 32)   // 256

__global__ __launch_bounds__(128) void vq_kernel(float* __restrict__ ws) {
  __shared__ __align__(16) __bf16 sB[2][32 * SB_STRIDE];
  const float*  __restrict__ pooled  = ws + WS_POOLED;
  const float*  __restrict__ embnorm = ws + WS_EMBNORM;
  const __bf16* __restrict__ ebf     = (const __bf16*)(ws + WS_EMBBF);
  int* __restrict__ idxp = (int*)(ws + WS_IDX);

  const int tid = threadIdx.x;
  const int lane = tid & 31;
  const int wid = tid >> 5;
  const int m0 = blockIdx.x * 64 + wid * 16;
  const int rlane = lane & 15;
  const int khalf = (lane >> 4) << 3;           // 0 or 8

  // --- A fragments: 16 rows x 512 dims, f32 -> bf16, held in registers.
  // ISA 16-bit A 16x32 layout: elems 0..7 = K{kb+khalf..}, 8..15 = K{kb+16+khalf..}
  const float* pr = pooled + (size_t)(m0 + rlane) * EMB_D;
  v16bf Areg[16];
#pragma unroll
  for (int kk = 0; kk < 16; ++kk) {
    const int c0 = kk * 32 + khalf;
    float4 a0 = *(const float4*)(pr + c0);
    float4 a1 = *(const float4*)(pr + c0 + 4);
    float4 a2 = *(const float4*)(pr + c0 + 16);
    float4 a3 = *(const float4*)(pr + c0 + 20);
    v16bf A;
    A[0]=(__bf16)a0.x;  A[1]=(__bf16)a0.y;  A[2]=(__bf16)a0.z;  A[3]=(__bf16)a0.w;
    A[4]=(__bf16)a1.x;  A[5]=(__bf16)a1.y;  A[6]=(__bf16)a1.z;  A[7]=(__bf16)a1.w;
    A[8]=(__bf16)a2.x;  A[9]=(__bf16)a2.y;  A[10]=(__bf16)a2.z; A[11]=(__bf16)a2.w;
    A[12]=(__bf16)a3.x; A[13]=(__bf16)a3.y; A[14]=(__bf16)a3.z; A[15]=(__bf16)a3.w;
    Areg[kk] = A;
  }

  float minv[8]; int mini[8];
#pragma unroll
  for (int r = 0; r < 8; ++r) { minv[r] = 3.402823466e38f; mini[r] = 0; }

  // tile fill: 32 code rows x 64 chunks of 8 bf16 (16B) -> 16 chunks/thread
  auto fill_tile = [&](int nt, int buf) {
    const __bf16* src = ebf + (size_t)nt * 32 * EMB_D;
    __bf16* dst = &sB[buf][0];
    for (int i = tid; i < 32 * 64; i += 128) {
      const int r = i >> 6;
      const int c = (i & 63) << 3;
#ifdef USE_ASYNC_LDS
      async_copy16(src + (size_t)r * EMB_D + c, dst + r * SB_STRIDE + c);
#else
      v8bf v = *(const v8bf*)(src + (size_t)r * EMB_D + c);
      *(v8bf*)(dst + r * SB_STRIDE + c) = v;
#endif
    }
  };

  fill_tile(0, 0);
  wait_async_lds();
  __syncthreads();

  for (int nt = 0; nt < NT_ITERS; ++nt) {
    const int cur = nt & 1;
    if (nt + 1 < NT_ITERS) fill_tile(nt + 1, cur ^ 1);   // overlap with compute

    const __bf16* br0 = &sB[cur][rlane * SB_STRIDE];
    const __bf16* br1 = &sB[cur][(16 + rlane) * SB_STRIDE];
    v8f acc0 = {}, acc1 = {};
#pragma unroll
    for (int kk = 0; kk < 16; ++kk) {
      const int c0 = kk * 32 + khalf;
      v8bf p0 = *(const v8bf*)(br0 + c0);
      v8bf p1 = *(const v8bf*)(br0 + c0 + 16);
      v8bf q0 = *(const v8bf*)(br1 + c0);
      v8bf q1 = *(const v8bf*)(br1 + c0 + 16);
      v16bf B0 = __builtin_shufflevector(p0, p1,
          0,1,2,3,4,5,6,7,8,9,10,11,12,13,14,15);
      v16bf B1 = __builtin_shufflevector(q0, q1,
          0,1,2,3,4,5,6,7,8,9,10,11,12,13,14,15);
      acc0 = __builtin_amdgcn_wmma_f32_16x16x32_bf16(
          false, Areg[kk], false, B0, (short)0, acc0, false, false);
      acc1 = __builtin_amdgcn_wmma_f32_16x16x32_bf16(
          false, Areg[kk], false, B1, (short)0, acc1, false, false);
    }
    // dist = ||e||^2 - 2<pooled,e>  (||pooled||^2 constant per row)
    const int code0 = nt * 32 + rlane;
    const int code1 = code0 + 16;
    const float en0 = embnorm[code0];
    const float en1 = embnorm[code1];
#pragma unroll
    for (int r = 0; r < 8; ++r) {
      float d0 = en0 - 2.0f * acc0[r];
      if (d0 < minv[r]) { minv[r] = d0; mini[r] = code0; }
      float d1 = en1 - 2.0f * acc1[r];
      if (d1 < minv[r]) { minv[r] = d1; mini[r] = code1; }
    }
    wait_async_lds();
    __syncthreads();
  }

  // cross-lane argmin within each 16-lane half (tie-break: lower index)
#pragma unroll
  for (int r = 0; r < 8; ++r) {
    float v = minv[r]; int i = mini[r];
#pragma unroll
    for (int m = 8; m >= 1; m >>= 1) {
      float ov = __shfl_xor(v, m, 32);
      int   oi = __shfl_xor(i, m, 32);
      if (ov < v || (ov == v && oi < i)) { v = ov; i = oi; }
    }
    if (rlane == 0) idxp[m0 + (lane >> 4) * 8 + r] = i;
  }
}

// ---------------------------------------------------------------------------
// Kernel 4: gather emb[idx], broadcast over span, e-latent partials, idx_out.
// grid = N_SEG blocks, 256 threads.
// ---------------------------------------------------------------------------
__global__ __launch_bounds__(256) void output_kernel(
    const float* __restrict__ emb, float* __restrict__ ws,
    float* __restrict__ out) {
  __shared__ float red[256];
  const int tid = threadIdx.x;
  const int n = blockIdx.x;
  const int b = n / S_LEN, s = n % S_LEN;
  const int d0 = tid * 2;
  const int idx = ((const int*)(ws + WS_IDX))[n];
  float2 q = *(const float2*)(emb + (size_t)idx * EMB_D + d0);
  float2 p = *(const float2*)(ws + WS_POOLED + (size_t)n * EMB_D + d0);
  const float dx = q.x - p.x, dy = q.y - p.y;
  red[tid] = dx * dx + dy * dy;
  const size_t obase = ((size_t)(b * T_LEN + s * SPAN_SZ)) * EMB_D + d0;
#pragma unroll
  for (int j = 0; j < SPAN_SZ; ++j)
    *(float2*)(out + OUT_Q + obase + (size_t)j * EMB_D) = q;
  __syncthreads();
  for (int off = 128; off >= 1; off >>= 1) {
    if (tid < off) red[tid] += red[tid + off];
    __syncthreads();
  }
  if (tid == 0) ws[WS_EPART + n] = red[0];
  if (tid < SPAN_SZ)
    out[OUT_IDX + (size_t)b * T_LEN + s * SPAN_SZ + tid] = (float)idx;
}

// ---------------------------------------------------------------------------
// Kernel 5: deterministic fixed-order reduction of partials -> total_loss
// ---------------------------------------------------------------------------
__global__ __launch_bounds__(256) void finalize_kernel(
    const float* __restrict__ ws, float* __restrict__ out) {
  __shared__ float rb[256], re[256];
  const int tid = threadIdx.x;
  float sb = 0.f, se = 0.f;
  for (int i = tid; i < N_SEG; i += 256) {
    sb += ws[WS_BPART + i];
    se += ws[WS_EPART + i];
  }
  rb[tid] = sb; re[tid] = se;
  __syncthreads();
  for (int off = 128; off >= 1; off >>= 1) {
    if (tid < off) { rb[tid] += rb[tid + off]; re[tid] += re[tid + off]; }
    __syncthreads();
  }
  if (tid == 0) {
    const float e_latent = re[0] / (float)((size_t)N_SEG * EMB_D);
    const float bmean = rb[0] / (float)BT;
    const float bl = bmean - 0.125f;   // target_rate = 1/SPAN
    out[OUT_LOSS] = 0.25f * e_latent + 0.01f * (bl * bl);
  }
}

extern "C" void kernel_launch(void* const* d_in, const int* in_sizes, int n_in,
                              void* d_out, int out_size, void* d_ws, size_t ws_size,
                              hipStream_t stream) {
  const float* x   = (const float*)d_in[0];
  const float* emb = (const float*)d_in[1];
  const float* Wb  = (const float*)d_in[2];
  const float* bb  = (const float*)d_in[3];
  float* out = (float*)d_out;
  float* ws  = (float*)d_ws;

  prep_kernel    <<<N_SEG,         256, 0, stream>>>(x, Wb, bb, out, ws);
  embnorm_kernel <<<NUM_EMB_K / 8, 256, 0, stream>>>(emb, ws);
  vq_kernel      <<<N_SEG / 64,    128, 0, stream>>>(ws);
  output_kernel  <<<N_SEG,         256, 0, stream>>>(emb, ws, out);
  finalize_kernel<<<1,             256, 0, stream>>>(ws, out);
}
